// MultiheadAttention_57647051047149
// MI455X (gfx1250) — matrix operand order repaired and checked by reading, not compiled, not run on hardware
//
#include <hip/hip_runtime.h>

#define D_MODEL 1024
#define NHEAD   16
#define HEAD_DIM 64
#define BATCH   2
#define SEQ     2048
#define MROWS   (BATCH * SEQ)   // 4096

typedef __attribute__((ext_vector_type(16))) __bf16 v16bf;
typedef __attribute__((ext_vector_type(8)))  __bf16 v8bf;
typedef __attribute__((ext_vector_type(8)))  float  v8f;
typedef __attribute__((ext_vector_type(4)))  unsigned short v4us;
typedef __attribute__((ext_vector_type(4)))  unsigned int   v4ui;
typedef __attribute__((ext_vector_type(8)))  int            v8i;
typedef __attribute__((ext_vector_type(4)))  int            v4i;

#if defined(__has_builtin)
#if __has_builtin(__builtin_amdgcn_tensor_load_to_lds) && \
    __has_builtin(__builtin_amdgcn_s_wait_tensorcnt)
#define USE_TDM 1
#endif
#endif
#ifndef USE_TDM
#define USE_TDM 0
#endif

union V16U { v16bf v; v8bf h[2]; };

__device__ __forceinline__ unsigned short f32_to_bf16_bits(float x) {
    unsigned int u = __float_as_uint(x);
    u += 0x7FFFu + ((u >> 16) & 1u);           // round-to-nearest-even
    return (unsigned short)(u >> 16);
}

__device__ __forceinline__ v8f wmma_bf16(v16bf a, v16bf b, v8f c) {
    return __builtin_amdgcn_wmma_f32_16x16x32_bf16(false, a, false, b,
                                                   (short)0, c, false, false);
}

#if USE_TDM
// Issue one 2-D TDM tile load (global -> LDS).  D# per CDNA5 ISA §8:
// group0: count=1 | lds_addr | global_addr[56:0] | type=2
// group1: data_size=2B, tensor_dim0/1, tile_dim0/1, tensor_dim0_stride
// This toolchain's builtin takes 6 args (g0 x4, g1 x8, g2 x4, g3 x4, x8, cpol).
__device__ __forceinline__ void tdm_load_2d(
    const void* gaddr, unsigned lds_off,
    unsigned tensor_d0, unsigned tensor_d1,
    unsigned tile_d0, unsigned tile_d1,
    unsigned long long stride0_elems)
{
    const unsigned long long ga = (unsigned long long)(size_t)gaddr;
    v4ui g0;
    g0.x = 1u;                                           // count=1 (user mode)
    g0.y = lds_off;                                      // LDS byte address
    g0.z = (unsigned)(ga & 0xFFFFFFFFu);                 // global_addr[31:0]
    g0.w = (unsigned)((ga >> 32) & 0x1FFFFFFu) | (2u << 30); // [56:32]|type=2
    v8i g1;
    g1[0] = (int)(1u << 16);                             // data_size=1 (2 bytes)
    g1[1] = (int)((tensor_d0 & 0xFFFFu) << 16);          // tensor_dim0[15:0]
    g1[2] = (int)(((tensor_d0 >> 16) & 0xFFFFu) |        // tensor_dim0[31:16]
                  ((tensor_d1 & 0xFFFFu) << 16));        // tensor_dim1[15:0]
    g1[3] = (int)(((tensor_d1 >> 16) & 0xFFFFu) |        // tensor_dim1[31:16]
                  ((tile_d0 & 0xFFFFu) << 16));          // tile_dim0
    g1[4] = (int)(tile_d1 & 0xFFFFu);                    // tile_dim1 (dim2=0)
    g1[5] = (int)(stride0_elems & 0xFFFFFFFFull);        // dim0 stride lo32
    g1[6] = (int)((stride0_elems >> 32) & 0xFFFFull);    // dim0 stride hi16
    g1[7] = 0;
    const v4i z4 = {0, 0, 0, 0};                         // 2-D: groups 2/3 unused
    const v8i z8 = {0, 0, 0, 0, 0, 0, 0, 0};
    __builtin_amdgcn_tensor_load_to_lds(g0, g1, z4, z4, z8, 0);
}
#endif

// ---------------------------------------------------------------- convert
__global__ __launch_bounds__(256) void cvt_f32_bf16(const float* __restrict__ src,
                                                    unsigned short* __restrict__ dst,
                                                    int n4) {
    int i = blockIdx.x * blockDim.x + threadIdx.x;
    int stride = gridDim.x * blockDim.x;
    for (; i < n4; i += stride) {
        const float4 f = ((const float4*)src)[i];
        v4us o;
        o.x = f32_to_bf16_bits(f.x);
        o.y = f32_to_bf16_bits(f.y);
        o.z = f32_to_bf16_bits(f.z);
        o.w = f32_to_bf16_bits(f.w);
        ((v4us*)dst)[i] = o;
    }
}

// ---------------------------------------------------------------- GEMM
// C[m,n] = sum_k A[m,k] * W[n,k] + bias[n]; 32x64 tile/wave, 8 WMMAs/k-step.
// mode 0: bf16 [b,h,s,d] | mode 2: bf16 [b,h,d,s] | mode 3: f32 row-major
__global__ __launch_bounds__(256) void gemm_bf16_wmma(
    const unsigned short* __restrict__ A,
    const unsigned short* __restrict__ W,
    const float* __restrict__ bias,
    unsigned short* __restrict__ out_bf,
    float* __restrict__ out_f32,
    int mode)
{
    const int K = D_MODEL;
    const int lane = threadIdx.x & 31;
    const int wave = threadIdx.x >> 5;
    const int half = lane >> 4;
    const int l16  = lane & 15;

    const int w  = blockIdx.x * 8 + wave;
    const int n0 = (w & 15) * 64;
    const int m0 = (w >> 4) * 32;

    const unsigned short* Arow[2];
    const unsigned short* Wrow[4];
#pragma unroll
    for (int i = 0; i < 2; ++i) Arow[i] = A + (size_t)(m0 + 16 * i + l16) * K;
#pragma unroll
    for (int j = 0; j < 4; ++j) Wrow[j] = W + (size_t)(n0 + 16 * j + l16) * K;

    v8f acc[2][4] = {};
    for (int k0 = 0; k0 < K; k0 += 32) {
        __builtin_prefetch(Arow[0] + k0 + 256, 0, 3);
        __builtin_prefetch(Wrow[0] + k0 + 256, 0, 3);
        V16U a[2], bf[4];
#pragma unroll
        for (int i = 0; i < 2; ++i) {
            a[i].h[0] = *(const v8bf*)(Arow[i] + k0 + 8 * half);
            a[i].h[1] = *(const v8bf*)(Arow[i] + k0 + 16 + 8 * half);
        }
#pragma unroll
        for (int j = 0; j < 4; ++j) {
            bf[j].h[0] = *(const v8bf*)(Wrow[j] + k0 + 8 * half);
            bf[j].h[1] = *(const v8bf*)(Wrow[j] + k0 + 16 + 8 * half);
        }
#pragma unroll
        for (int i = 0; i < 2; ++i)
#pragma unroll
            for (int j = 0; j < 4; ++j)
                acc[i][j] = wmma_bf16(a[i].v, bf[j].v, acc[i][j]);
    }

    float bvn[4];
#pragma unroll
    for (int j = 0; j < 4; ++j) bvn[j] = bias[n0 + 16 * j + l16];

#pragma unroll
    for (int i = 0; i < 2; ++i)
#pragma unroll
        for (int j = 0; j < 4; ++j)
#pragma unroll
            for (int r = 0; r < 8; ++r) {
                const float val = acc[i][j][r] + bvn[j];
                const int m = m0 + 16 * i + r + 8 * half;
                const int n = n0 + 16 * j + l16;
                const int bb = m >> 11;
                const int s  = m & (SEQ - 1);
                const int h  = n >> 6;
                const int d  = n & (HEAD_DIM - 1);
                if (mode == 0) {
                    out_bf[(((size_t)(bb * NHEAD + h) * SEQ) + s) * HEAD_DIM + d] =
                        f32_to_bf16_bits(val);
                } else if (mode == 2) {
                    out_bf[(((size_t)(bb * NHEAD + h) * HEAD_DIM) + d) * SEQ + s] =
                        f32_to_bf16_bits(val);
                } else {
                    out_f32[(size_t)m * D_MODEL + n] = val;
                }
            }
}

// ---------------------------------------------------------------- attention
// Per-chunk compute body shared by the TDM and direct-load paths.
// Kc: 32 key rows, row stride krs; Vc: 64 dim rows, row stride vrs.
__device__ __forceinline__ void attn_chunk_body(
    const unsigned short* __restrict__ Kc, const int krs,
    const unsigned short* __restrict__ Vc, const int vrs,
    const V16U (&qf)[2][2], v8f (&o)[2][4],
    float (&mrun)[2][8], float (&lrun)[2][8],
    unsigned short* __restrict__ P, const int half, const int l16)
{
    V16U bsc[2][2];
#pragma unroll
    for (int sub = 0; sub < 2; ++sub) {
        const unsigned short* Krow = Kc + (size_t)(16 * sub + l16) * krs;
        bsc[sub][0].h[0] = *(const v8bf*)(Krow + 8 * half);
        bsc[sub][0].h[1] = *(const v8bf*)(Krow + 16 + 8 * half);
        bsc[sub][1].h[0] = *(const v8bf*)(Krow + 32 + 8 * half);
        bsc[sub][1].h[1] = *(const v8bf*)(Krow + 48 + 8 * half);
    }

    v8f sc[2][2];
#pragma unroll
    for (int i = 0; i < 2; ++i)
#pragma unroll
        for (int sub = 0; sub < 2; ++sub) {
            v8f acc = {};
            acc = wmma_bf16(qf[i][0].v, bsc[sub][0].v, acc);
            acc = wmma_bf16(qf[i][1].v, bsc[sub][1].v, acc);
            sc[i][sub] = acc * 0.125f;            // 1/sqrt(64)
        }

#pragma unroll
    for (int i = 0; i < 2; ++i) {
        float alpha[8];
#pragma unroll
        for (int r = 0; r < 8; ++r) {
            float t = fmaxf(sc[i][0][r], sc[i][1][r]);
            t = fmaxf(t, __shfl_xor(t, 1, 32));
            t = fmaxf(t, __shfl_xor(t, 2, 32));
            t = fmaxf(t, __shfl_xor(t, 4, 32));
            t = fmaxf(t, __shfl_xor(t, 8, 32));
            const float mn = fmaxf(mrun[i][r], t);
            alpha[r] = __expf(mrun[i][r] - mn);
            mrun[i][r] = mn;
            const float p0 = __expf(sc[i][0][r] - mn);
            const float p1 = __expf(sc[i][1][r] - mn);
            sc[i][0][r] = p0; sc[i][1][r] = p1;
            float cs = p0 + p1;
            cs += __shfl_xor(cs, 1, 32);
            cs += __shfl_xor(cs, 2, 32);
            cs += __shfl_xor(cs, 4, 32);
            cs += __shfl_xor(cs, 8, 32);
            lrun[i][r] = lrun[i][r] * alpha[r] + cs;
        }
#pragma unroll
        for (int t = 0; t < 4; ++t)
#pragma unroll
            for (int r = 0; r < 8; ++r) o[i][t][r] *= alpha[r];

#pragma unroll
        for (int r = 0; r < 8; ++r) {
            const int row = 16 * i + r + 8 * half;
            P[row * 32 + l16]      = f32_to_bf16_bits(sc[i][0][r]);
            P[row * 32 + 16 + l16] = f32_to_bf16_bits(sc[i][1][r]);
        }
    }
    asm volatile("s_wait_dscnt 0" ::: "memory");

    V16U pf[2];
#pragma unroll
    for (int i = 0; i < 2; ++i) {
        const unsigned short* Prow = P + (16 * i + l16) * 32;
        pf[i].h[0] = *(const v8bf*)(Prow + 8 * half);
        pf[i].h[1] = *(const v8bf*)(Prow + 16 + 8 * half);
    }

    V16U vb[4];
#pragma unroll
    for (int t = 0; t < 4; ++t) {
        const unsigned short* Vrow = Vc + (size_t)(16 * t + l16) * vrs;
        vb[t].h[0] = *(const v8bf*)(Vrow + 8 * half);
        vb[t].h[1] = *(const v8bf*)(Vrow + 16 + 8 * half);
    }
#pragma unroll
    for (int i = 0; i < 2; ++i)
#pragma unroll
        for (int t = 0; t < 4; ++t)
            o[i][t] = wmma_bf16(pf[i].v, vb[t].v, o[i][t]);
}

__global__ __launch_bounds__(256) void attn_wmma(
    const unsigned short* __restrict__ Qb,    // [B,H,S,Dh] bf16
    const unsigned short* __restrict__ Kb,    // [B,H,S,Dh] bf16
    const unsigned short* __restrict__ Vt,    // [B,H,Dh,S] bf16
    unsigned short* __restrict__ attn)        // [B,S,D_MODEL] bf16
{
    __shared__ unsigned short ldsP[8 * 32 * 32];       // P tiles, 16 KB
#if USE_TDM
    extern __shared__ unsigned short ldsStage[];       // 128 KB staging
#endif

    const int lane = threadIdx.x & 31;
    const int wave = threadIdx.x >> 5;
    const int half = lane >> 4;
    const int l16  = lane & 15;
    const int job  = blockIdx.x * 8 + wave;     // 2048 jobs
    const int bh   = job >> 6;
    const int qt   = job & 63;
    const int b    = bh >> 4;
    const int h    = bh & 15;
    const int s0   = qt * 32;

    const unsigned short* Kbase = Kb + (size_t)bh * SEQ * HEAD_DIM;
    const unsigned short* Vbase = Vt + (size_t)bh * HEAD_DIM * SEQ;
    unsigned short* P = ldsP + wave * (32 * 32);

    V16U qf[2][2];
#pragma unroll
    for (int i = 0; i < 2; ++i) {
        const unsigned short* Qrow =
            Qb + ((size_t)bh * SEQ + s0 + 16 * i + l16) * HEAD_DIM;
        qf[i][0].h[0] = *(const v8bf*)(Qrow + 8 * half);
        qf[i][0].h[1] = *(const v8bf*)(Qrow + 16 + 8 * half);
        qf[i][1].h[0] = *(const v8bf*)(Qrow + 32 + 8 * half);
        qf[i][1].h[1] = *(const v8bf*)(Qrow + 48 + 8 * half);
    }

    v8f o[2][4] = {};
    float mrun[2][8], lrun[2][8];
#pragma unroll
    for (int i = 0; i < 2; ++i)
#pragma unroll
        for (int r = 0; r < 8; ++r) { mrun[i][r] = -3.0e38f; lrun[i][r] = 0.0f; }

    const int NCH = SEQ / 32;   // 64 key chunks

#if USE_TDM
    // per-wave staging: [K0|K1|V0|V1] x 4KB (2048 bf16 each)
    unsigned short* stage = ldsStage + wave * 8192;
    const unsigned kOffLds[2] = { (unsigned)(size_t)(stage),
                                  (unsigned)(size_t)(stage + 2048) };
    const unsigned vOffLds[2] = { (unsigned)(size_t)(stage + 4096),
                                  (unsigned)(size_t)(stage + 6144) };
    // prologue: DMA chunk 0 into buffer 0 (K tile 32x64, Vt tile 64x32)
    tdm_load_2d(Kbase, kOffLds[0], HEAD_DIM, SEQ, HEAD_DIM, 32, HEAD_DIM);
    tdm_load_2d(Vbase, vOffLds[0], SEQ, HEAD_DIM, 32, HEAD_DIM, SEQ);

    for (int c = 0; c < NCH; ++c) {
        const int cur = c & 1;
        if (c + 1 < NCH) {
            const int nxt = (c + 1) & 1;
            tdm_load_2d(Kbase + (size_t)(c + 1) * 32 * HEAD_DIM, kOffLds[nxt],
                        HEAD_DIM, SEQ, HEAD_DIM, 32, HEAD_DIM);
            tdm_load_2d(Vbase + (size_t)(c + 1) * 32, vOffLds[nxt],
                        SEQ, HEAD_DIM, 32, HEAD_DIM, SEQ);
            __builtin_amdgcn_s_wait_tensorcnt(2);   // current chunk landed
        } else {
            __builtin_amdgcn_s_wait_tensorcnt(0);
        }
        const unsigned short* Kc = stage + cur * 2048;         // [32][64]
        const unsigned short* Vc = stage + 4096 + cur * 2048;  // [64][32]
        attn_chunk_body(Kc, HEAD_DIM, Vc, 32,
                        qf, o, mrun, lrun, P, half, l16);
    }
#else
    for (int c = 0; c < NCH; ++c) {
        const int n0 = c * 32;
        attn_chunk_body(Kbase + (size_t)n0 * HEAD_DIM, HEAD_DIM,
                        Vbase + n0, SEQ,
                        qf, o, mrun, lrun, P, half, l16);
    }
#endif

    // epilogue: normalize, store as [b, s, h*Dh+d] bf16
#pragma unroll
    for (int i = 0; i < 2; ++i) {
        float inv[8];
#pragma unroll
        for (int r = 0; r < 8; ++r) inv[r] = 1.0f / lrun[i][r];
#pragma unroll
        for (int t = 0; t < 4; ++t)
#pragma unroll
            for (int r = 0; r < 8; ++r) {
                const int row = s0 + 16 * i + r + 8 * half;
                const int d   = h * HEAD_DIM + 16 * t + l16;
                attn[((size_t)b * SEQ + row) * D_MODEL + d] =
                    f32_to_bf16_bits(o[i][t][r] * inv[r]);
            }
    }
}

// ---------------------------------------------------------------- launch
extern "C" void kernel_launch(void* const* d_in, const int* in_sizes, int n_in,
                              void* d_out, int out_size, void* d_ws, size_t ws_size,
                              hipStream_t stream) {
    const float* query = (const float*)d_in[0];
    const float* key   = (const float*)d_in[1];
    const float* value = (const float*)d_in[2];
    const float* Wq = (const float*)d_in[3];
    const float* bq = (const float*)d_in[4];
    const float* Wk = (const float*)d_in[5];
    const float* bk = (const float*)d_in[6];
    const float* Wv = (const float*)d_in[7];
    const float* bv = (const float*)d_in[8];
    const float* Wo = (const float*)d_in[9];
    const float* bo = (const float*)d_in[10];

    const size_t NTOK = (size_t)MROWS * D_MODEL;     // 4M elems
    const size_t WSZ  = (size_t)D_MODEL * D_MODEL;   // 1M elems

    unsigned short* ws    = (unsigned short*)d_ws;
    unsigned short* q_bf  = ws;
    unsigned short* k_bf  = q_bf + NTOK;
    unsigned short* v_bf  = k_bf + NTOK;
    unsigned short* wq_bf = v_bf + NTOK;
    unsigned short* wk_bf = wq_bf + WSZ;
    unsigned short* wv_bf = wk_bf + WSZ;
    unsigned short* wo_bf = wv_bf + WSZ;
    unsigned short* Qbuf  = wo_bf + WSZ;
    unsigned short* Kbuf  = Qbuf + NTOK;
    unsigned short* Vtb   = Kbuf + NTOK;
    unsigned short* attnb = Vtb + NTOK;   // total 32M bf16 = 64 MB

    const dim3 cblk(256), cgrd(1024);
    cvt_f32_bf16<<<cgrd, cblk, 0, stream>>>(query, q_bf, (int)(NTOK / 4));
    cvt_f32_bf16<<<cgrd, cblk, 0, stream>>>(key,   k_bf, (int)(NTOK / 4));
    cvt_f32_bf16<<<cgrd, cblk, 0, stream>>>(value, v_bf, (int)(NTOK / 4));
    cvt_f32_bf16<<<cgrd, cblk, 0, stream>>>(Wq, wq_bf, (int)(WSZ / 4));
    cvt_f32_bf16<<<cgrd, cblk, 0, stream>>>(Wk, wk_bf, (int)(WSZ / 4));
    cvt_f32_bf16<<<cgrd, cblk, 0, stream>>>(Wv, wv_bf, (int)(WSZ / 4));
    cvt_f32_bf16<<<cgrd, cblk, 0, stream>>>(Wo, wo_bf, (int)(WSZ / 4));

    const dim3 gblk(256), ggrd((MROWS / 32) * (D_MODEL / 64) / 8);  // 256
    gemm_bf16_wmma<<<ggrd, gblk, 0, stream>>>(q_bf, wq_bf, bq, Qbuf, nullptr, 0);
    gemm_bf16_wmma<<<ggrd, gblk, 0, stream>>>(k_bf, wk_bf, bk, Kbuf, nullptr, 0);
    gemm_bf16_wmma<<<ggrd, gblk, 0, stream>>>(v_bf, wv_bf, bv, Vtb,  nullptr, 2);

    // flash attention, 8 waves/block; 128KB dynamic LDS for TDM staging
    attn_wmma<<<dim3(256), dim3(256), 128 * 1024, stream>>>(Qbuf, Kbuf, Vtb, attnb);

    gemm_bf16_wmma<<<ggrd, gblk, 0, stream>>>(attnb, wo_bf, bo, nullptr,
                                              (float*)d_out, 3);
}
